// ScaledDotProductAttention_81801947120246
// MI455X (gfx1250) — compile-verified, matrix operand order checked by reading, not run on hardware
//
#include <hip/hip_runtime.h>
#include <hip/hip_bf16.h>
#include <math.h>

// ---------------------------------------------------------------------------
// ScaledDotProductAttention for MI455X (gfx1250, wave32, WMMA)
// B=16, Sq=Sk=2048, Dk=Dv=128, fp32 in/out; outputs context + attn_weights.
//
// Roofline: mandatory HBM traffic ~0.59 GB (268 MB mask read + 268 MB weight
// write + ~50 MB QKV/ctx) -> ~25 us @ 23.3 TB/s. Matmuls (34.4 GFLOP) use
// v_wmma_f32_16x16x32_bf16 so compute sits far below that floor.
//
// One workgroup (4 waves) owns a 16-row Q stripe; the full 16x2048 fp32 score
// stripe lives in LDS (128 KB of 320 KB/WGP -> 2 WGs per WGP). Softmax phases
// are wave-private (4 rows per wave), 128-bit vectorized, and use NT cache
// hints so the one-touch mask/weight streams don't evict K/V from the 192 MB
// L2 (K/V: 32 MB total, reused by 128 WGs per batch).
// ---------------------------------------------------------------------------

#define SQK 2048
#define DKV 128

typedef __attribute__((ext_vector_type(16))) __bf16 v16bf;
typedef __attribute__((ext_vector_type(8)))  float  v8f;
typedef __attribute__((ext_vector_type(4)))  float  f32x4;
typedef __attribute__((ext_vector_type(4)))  int    i32x4;

// 16x32 fp32 tile (row-major, leading dim ld) -> CDNA5 16-bit A/B fragment:
//   lanes 0-15 : row M = lane,    halves 0-7 -> K=0..7,  halves 8-15 -> K=16..23
//   lanes16-31 : row M = lane-16, halves 0-7 -> K=8..15, halves 8-15 -> K=24..31
__device__ __forceinline__ v16bf load_frag_rows_f32(const float* base, int ld) {
  const int lane = threadIdx.x & 31;
  const int row  = lane & 15;
  const int sel  = lane >> 4;
  const float* p = base + (size_t)row * ld + sel * 8;
  v16bf r;
#pragma unroll
  for (int i = 0; i < 8; ++i) r[i] = (__bf16)p[i];
#pragma unroll
  for (int i = 0; i < 8; ++i) r[8 + i] = (__bf16)p[16 + i];
  return r;
}

// 32x16 fp32 tile (element [kk][n] = base[kk*ld + n]) -> B fragment
// (lane = column N; K split exactly as above).
__device__ __forceinline__ v16bf load_frag_cols_f32(const float* base, int ld) {
  const int lane = threadIdx.x & 31;
  const int n    = lane & 15;
  const int sel  = lane >> 4;
  v16bf r;
#pragma unroll
  for (int h = 0; h < 8; ++h)
    r[h] = (__bf16)base[(size_t)(sel * 8 + h) * ld + n];
#pragma unroll
  for (int h = 0; h < 8; ++h)
    r[8 + h] = (__bf16)base[(size_t)(16 + sel * 8 + h) * ld + n];
  return r;
}

__device__ __forceinline__ float wave_reduce_max(float v) {
#pragma unroll
  for (int off = 16; off > 0; off >>= 1)
    v = fmaxf(v, __shfl_xor(v, off, 32));
  return v;
}

__device__ __forceinline__ float wave_reduce_sum(float v) {
#pragma unroll
  for (int off = 16; off > 0; off >>= 1)
    v += __shfl_xor(v, off, 32);
  return v;
}

__global__ void sdpa_fused_kernel(const float* __restrict__ Q,
                                  const float* __restrict__ K,
                                  const float* __restrict__ V,
                                  const int*   __restrict__ mask,
                                  float* __restrict__ ctx_out,
                                  float* __restrict__ w_out) {
  extern __shared__ float lds[];
  float* S = lds;                      // 16 x 2048 fp32 score stripe (128 KB)

  const int wg   = blockIdx.x;         // 0 .. B*(SQ/16)-1
  const int b    = wg >> 7;            // / (2048/16)
  const int q0   = (wg & 127) << 4;    // * 16
  const int tid  = threadIdx.x;
  const int wave = tid >> 5;
  const int lane = tid & 31;

  const float* Qb = Q + ((size_t)b * SQK + q0) * DKV;
  const float* Kb = K + (size_t)b * SQK * DKV;
  const float* Vb = V + (size_t)b * SQK * DKV;
  const int*   Mb = mask + ((size_t)b * SQK + q0) * (size_t)SQK;

  // ---- Preload Q fragments covering all of Dk=128 (4 x K=32). -------------
  v16bf qf[4];
#pragma unroll
  for (int i = 0; i < 4; ++i) qf[i] = load_frag_rows_f32(Qb + 32 * i, DKV);

  const float scale = 0.08838834764831845f;  // 1/sqrt(128)

  // ---- Phase 1: S = (Q K^T) * scale, streamed into LDS. -------------------
  // Wave w computes 16x16 tiles at columns k0 = w*16 + 64*j.
  for (int k0 = wave * 16; k0 < SQK; k0 += 64) {
    if (k0 + 64 < SQK)
      __builtin_prefetch(Kb + (size_t)(k0 + 64) * DKV + lane * 16, 0, 1);

    v8f acc = {};
#pragma unroll
    for (int i = 0; i < 4; ++i) {
      v16bf kf = load_frag_rows_f32(Kb + (size_t)k0 * DKV + 32 * i, DKV);
      acc = __builtin_amdgcn_wmma_f32_16x16x32_bf16(
          false, qf[i], false, kf, (short)0, acc, false, false);
    }
    const int n = lane & 15, sel = lane >> 4;
#pragma unroll
    for (int r = 0; r < 8; ++r)
      S[(size_t)(r + 8 * sel) * SQK + k0 + n] = acc[r] * scale;
  }
  __syncthreads();

  // ---- Phases 2+3: softmax, wave-private (wave owns rows wave*4..+3). -----
  // 128-bit lane accesses; mask load + weight store are non-temporal so the
  // one-touch streams don't evict K/V from L2.
  for (int rr = 0; rr < 4; ++rr) {
    const int r = wave * 4 + rr;
    f32x4*       Srow = (f32x4*)(S + (size_t)r * SQK);
    const i32x4* mrow = (const i32x4*)(Mb + (size_t)r * SQK);
    f32x4*       wrow =
        (f32x4*)(w_out + ((size_t)b * SQK + q0 + r) * (size_t)SQK);

    // Row max (pre-mask, as in the reference).
    float m = -INFINITY;
    for (int j = lane; j < SQK / 4; j += 32) {
      f32x4 v = Srow[j];
      m = fmaxf(m, fmaxf(fmaxf(v[0], v[1]), fmaxf(v[2], v[3])));
    }
    m = wave_reduce_max(m);

    // exp(S - max) * (mask != 0), accumulate row sum, keep in LDS.
    float s = 0.f;
    for (int j = lane; j < SQK / 4; j += 32) {
      f32x4 v  = Srow[j];
      i32x4 mk = __builtin_nontemporal_load(&mrow[j]);
      f32x4 w;
#pragma unroll
      for (int c = 0; c < 4; ++c)
        w[c] = (mk[c] != 0) ? __expf(v[c] - m) : 0.f;
      Srow[j] = w;
      s += (w[0] + w[1]) + (w[2] + w[3]);
    }
    s = wave_reduce_sum(s);
    const float inv = 1.0f / s;

    // Normalize: emit weights (NT) and keep normalized P in LDS for phase 4.
    for (int j = lane; j < SQK / 4; j += 32) {
      f32x4 w = Srow[j] * inv;
      Srow[j] = w;
      __builtin_nontemporal_store(w, &wrow[j]);
    }
  }
  __syncthreads();

  // ---- Phase 4: context = P V. Wave w owns output cols [w*32, w*32+32). ---
  {
    const int n0 = wave * 32;
    v8f acc0 = {}, acc1 = {};
    for (int k0 = 0; k0 < SQK; k0 += 32) {
      v16bf pf = load_frag_rows_f32(S + k0, SQK);
      v16bf v0 = load_frag_cols_f32(Vb + (size_t)k0 * DKV + n0, DKV);
      v16bf v1 = load_frag_cols_f32(Vb + (size_t)k0 * DKV + n0 + 16, DKV);
      acc0 = __builtin_amdgcn_wmma_f32_16x16x32_bf16(
          false, pf, false, v0, (short)0, acc0, false, false);
      acc1 = __builtin_amdgcn_wmma_f32_16x16x32_bf16(
          false, pf, false, v1, (short)0, acc1, false, false);
    }
    const int n = lane & 15, sel = lane >> 4;
#pragma unroll
    for (int r = 0; r < 8; ++r) {
      const int M = r + 8 * sel;
      __builtin_nontemporal_store(
          acc0[r], &ctx_out[((size_t)b * SQK + q0 + M) * DKV + n0 + n]);
      __builtin_nontemporal_store(
          acc1[r], &ctx_out[((size_t)b * SQK + q0 + M) * DKV + n0 + 16 + n]);
    }
  }
}

extern "C" void kernel_launch(void* const* d_in, const int* in_sizes, int n_in,
                              void* d_out, int out_size, void* d_ws, size_t ws_size,
                              hipStream_t stream) {
  (void)in_sizes; (void)n_in; (void)d_ws; (void)ws_size; (void)out_size;
  const float* Q    = (const float*)d_in[0];
  const float* K    = (const float*)d_in[1];
  const float* V    = (const float*)d_in[2];
  const int*   mask = (const int*)d_in[3];

  const int B = 16;
  float* ctx = (float*)d_out;                          // B*SQ*DV
  float* wts = (float*)d_out + (size_t)B * SQK * DKV;  // B*SQ*SK

  const int blocks  = B * (SQK / 16);      // 2048 workgroups
  const int threads = 128;                 // 4 waves (wave32)
  const size_t shmem = (size_t)(16 * SQK) * sizeof(float);  // 128 KB

  sdpa_fused_kernel<<<blocks, threads, shmem, stream>>>(Q, K, V, mask, ctx, wts);
}
